// MyTransformerBlock_17033840296423
// MI455X (gfx1250) — compile-verified, matrix operand order checked by reading, not run on hardware
//
#include <hip/hip_runtime.h>

// ---------------------------------------------------------------------------
// MyTransformerBlock on gfx1250 (MI455X): bf16 WMMA everywhere, fp32 accum.
// SEQ=2048, BS=2, DIM=256, G=2 kv-heads, NQ=8 queries (SUB=4 per group).
// GEMM weight tiles are staged to LDS by the Tensor Data Mover
// (tensor_load_to_lds + s_wait_tensorcnt), double-buffered per workgroup.
// ---------------------------------------------------------------------------

#define SEQ   2048
#define BSZ   2
#define DIM   256
#define NTOK  (SEQ * BSZ)      // 4096 token rows, t = s*BSZ + b
#define GK    2                // kv groups
#define NQH   8                // query heads per batch elem (g*4+u)
#define EPS   1e-5f

typedef __attribute__((ext_vector_type(16))) __bf16 bf16x16;
typedef __attribute__((ext_vector_type(8)))  __bf16 bf16x8;
typedef __attribute__((ext_vector_type(8)))  float  f32x8;
typedef __attribute__((ext_vector_type(4)))  unsigned int u32x4;
typedef __attribute__((ext_vector_type(8)))  int i32x8;
typedef __attribute__((ext_vector_type(4)))  int i32x4;

// Load one 16x32 bf16 A/B fragment from a row-major matrix (ld in elements).
// Per CDNA5 ISA 7.12.2: lanes 0-15 = rows, hold K {k0..k0+7, k0+16..k0+23};
// lanes 16-31 = same rows, hold K {k0+8..k0+15, k0+24..k0+31}.
__device__ __forceinline__ bf16x16 load_frag(const __bf16* p, int row0, int ld,
                                             int k0, int lane) {
    int r  = row0 + (lane & 15);
    int kb = k0 + ((lane >> 4) << 3);
    const __bf16* base = p + (size_t)r * ld + kb;
    bf16x8 lo = *reinterpret_cast<const bf16x8*>(base);
    bf16x8 hi = *reinterpret_cast<const bf16x8*>(base + 16);
    bf16x16 out;
#pragma unroll
    for (int i = 0; i < 8; ++i) { out[i] = lo[i]; out[i + 8] = hi[i]; }
    return out;
}

__device__ __forceinline__ f32x8 wmma_bf16(bf16x16 a, bf16x16 b, f32x8 c) {
    return __builtin_amdgcn_wmma_f32_16x16x32_bf16(false, a, false, b,
                                                   (short)0, c, false, false);
}

// Generic shared pointer -> LDS byte offset (generic LDS addr low 32 bits).
__device__ __forceinline__ unsigned lds_off(const void* p) {
    return (unsigned)(unsigned long long)(uintptr_t)p;
}

// TDM 2D tile load: copy tile_n rows x tile_k bf16 elements (row stride
// ld_elems) from global 'g' into contiguous LDS at 'lds_byte' ([n][k], ld=tile_k).
// D# per CDNA5 ISA 8.3/8.4: group0 {count=1, lds_addr, global_addr, type=2},
// group1 {data_size=2B, tensor dims = tile dims, tile_dim0/1, dim0 stride}.
__device__ __forceinline__ void tdm_load_tile(unsigned lds_byte, const void* g,
                                              unsigned tile_k, unsigned tile_n,
                                              unsigned ld_elems) {
    unsigned long long ga = (unsigned long long)(uintptr_t)g;
    u32x4 g0 = {};
    g0[0] = 1u;                                      // count=1, user descriptor
    g0[1] = lds_byte;                                // lds_addr
    g0[2] = (unsigned)(ga & 0xFFFFFFFFu);            // global_addr[31:0]
    g0[3] = (unsigned)((ga >> 32) & 0x01FFFFFFu)     // global_addr[56:32]
          | (2u << 30);                              // type=2 ("image")
    i32x8 g1 = {};
    g1[0] = 0x00010000;                              // data_size=1 -> 2 bytes
    g1[1] = (int)((tile_k & 0xFFFFu) << 16);         // tensor_dim0[15:0]
    g1[2] = (int)(((tile_k >> 16) & 0xFFFFu)         // tensor_dim0[31:16]
          | ((tile_n & 0xFFFFu) << 16));             // tensor_dim1[15:0]
    g1[3] = (int)(((tile_n >> 16) & 0xFFFFu)         // tensor_dim1[31:16]
          | ((tile_k & 0xFFFFu) << 16));             // tile_dim0
    g1[4] = (int)(tile_n & 0xFFFFu);                 // tile_dim1 (tile_dim2=0)
    g1[5] = (int)ld_elems;                           // tensor_dim0_stride[31:0]
    g1[6] = 0;                                       // stride[47:32], d1stride lo
    g1[7] = 0;
    i32x4 z4 = {};
#if defined(__clang_major__) && __clang_major__ >= 23
    i32x8 z8 = {};
    __builtin_amdgcn_tensor_load_to_lds(g0, g1, z4, z4, z8, 0);
#else
    __builtin_amdgcn_tensor_load_to_lds(g0, g1, z4, z4, 0);
#endif
}

// ---------------------------------------------------------------------------
// fp32 -> bf16 weight conversion
// ---------------------------------------------------------------------------
__global__ void cvt_bf16(const float* __restrict__ in, __bf16* __restrict__ out,
                         int n) {
    int i = blockIdx.x * blockDim.x + threadIdx.x;
    if (i < n) out[i] = (__bf16)in[i];
}

// ---------------------------------------------------------------------------
// Row layernorm over DIM=256: fp32 in -> bf16 out. One 256-thread block / row.
// ---------------------------------------------------------------------------
__global__ __launch_bounds__(256)
void ln_rows(const float* __restrict__ in, __bf16* __restrict__ out,
             const float* __restrict__ gamma, const float* __restrict__ beta) {
    __shared__ float red[256];
    int row = blockIdx.x, tid = threadIdx.x;
    float v = in[(size_t)row * DIM + tid];
    red[tid] = v; __syncthreads();
    for (int st = 128; st > 0; st >>= 1) {
        if (tid < st) red[tid] += red[tid + st];
        __syncthreads();
    }
    float mean = red[0] * (1.f / DIM);
    __syncthreads();
    float d = v - mean;
    red[tid] = d * d; __syncthreads();
    for (int st = 128; st > 0; st >>= 1) {
        if (tid < st) red[tid] += red[tid + st];
        __syncthreads();
    }
    float var = red[0] * (1.f / DIM);
    out[(size_t)row * DIM + tid] =
        (__bf16)(d * rsqrtf(var + EPS) * gamma[tid] + beta[tid]);
}

// ---------------------------------------------------------------------------
// Per-head post-processing of projection outputs (bf16 rows [NTOK, Gx*256]):
// optional layernorm (gn/bn), permutation to [head = b*Gx+g][s][256], or
// transposed [head][dim][s] (for V, so PV B-operands load row-major).
// ---------------------------------------------------------------------------
__global__ __launch_bounds__(256)
void heads_postln(const __bf16* __restrict__ in, __bf16* __restrict__ out,
                  const float* __restrict__ gamma, const float* __restrict__ beta,
                  int Gx, int do_ln, int transpose) {
    __shared__ float red[256];
    int tid = threadIdx.x;
    int g = blockIdx.x % Gx;
    int t = blockIdx.x / Gx;
    int bb = t & 1, s = t >> 1;
    float v = (float)in[(size_t)t * (Gx * DIM) + g * DIM + tid];
    float outv = v;
    if (do_ln) {
        red[tid] = v; __syncthreads();
        for (int st = 128; st > 0; st >>= 1) {
            if (tid < st) red[tid] += red[tid + st];
            __syncthreads();
        }
        float mean = red[0] * (1.f / DIM);
        __syncthreads();
        float d = v - mean;
        red[tid] = d * d; __syncthreads();
        for (int st = 128; st > 0; st >>= 1) {
            if (tid < st) red[tid] += red[tid + st];
            __syncthreads();
        }
        float var = red[0] * (1.f / DIM);
        outv = d * rsqrtf(var + EPS) * gamma[tid] + beta[tid];
    }
    int head = bb * Gx + g;
    size_t idx = transpose
        ? ((size_t)head * DIM * SEQ + (size_t)tid * SEQ + s)
        : ((size_t)head * SEQ * DIM + (size_t)s * DIM + tid);
    out[idx] = (__bf16)outv;
}

// ---------------------------------------------------------------------------
// Generic WMMA GEMM: C[M,N] = A[M,K](bf16) @ W[N,K]^T(bf16) + bias (+resid).
// 8 waves/WG; block tile 128(M)x64(N); the shared 64x32 W tile is DMA'd to
// LDS by the TDM (wave 0 issues tensor_load_to_lds, double-buffered,
// s_wait_tensorcnt ping-pong), giving 8x reuse of B fragments from LDS.
// M%128==0, N%64==0, K%32==0 for all call sites.
// ---------------------------------------------------------------------------
__global__ __launch_bounds__(256)
void gemm_wmma(const __bf16* __restrict__ A, int lda,
               const __bf16* __restrict__ W, int ldw,
               const float* __restrict__ bias,
               const float* __restrict__ resid,
               float* __restrict__ Cf, __bf16* __restrict__ Cb,
               int M, int N, int K, int relu) {
    __shared__ __bf16 sW[2][64 * 32];                 // 8 KB double buffer
    int lane = threadIdx.x & 31;
    int wave = threadIdx.x >> 5;
    int m0 = blockIdx.y * 128 + wave * 16;
    int n0 = blockIdx.x * 64;

    if (wave == 0)                                    // prologue: stage k0=0
        tdm_load_tile(lds_off(&sW[0][0]), W + (size_t)n0 * ldw, 32, 64, ldw);

    f32x8 acc[4] = {};
    int buf = 0;
    for (int k0 = 0; k0 < K; k0 += 32, buf ^= 1) {
        if (wave == 0) {
            if (k0 + 32 < K) {                        // overlap next stage
                tdm_load_tile(lds_off(&sW[buf ^ 1][0]),
                              W + (size_t)n0 * ldw + (k0 + 32), 32, 64, ldw);
                __builtin_amdgcn_s_wait_tensorcnt(1); // current buf done
            } else {
                __builtin_amdgcn_s_wait_tensorcnt(0);
            }
        }
        __syncthreads();                              // sW[buf] visible to all
        bf16x16 a = load_frag(A, m0, lda, k0, lane);
#pragma unroll
        for (int f = 0; f < 4; ++f) {
            bf16x16 b = load_frag(&sW[buf][0], 16 * f, 32, 0, lane);
            acc[f] = wmma_bf16(a, b, acc[f]);
        }
        __syncthreads();                              // reads done before reuse
    }

    int hf = lane >> 4, nn = lane & 15;
#pragma unroll
    for (int f = 0; f < 4; ++f) {
        int col = n0 + 16 * f + nn;
        float bv = bias ? bias[col] : 0.f;
#pragma unroll
        for (int r = 0; r < 8; ++r) {
            int row = m0 + r + 8 * hf;
            float v = acc[f][r] + bv;
            if (relu) v = fmaxf(v, 0.f);
            if (resid) v += resid[(size_t)row * N + col];
            if (Cf) Cf[(size_t)row * N + col] = v;
            if (Cb) Cb[(size_t)row * N + col] = (__bf16)v;
        }
    }
}

// ---------------------------------------------------------------------------
// Causal flash attention. Grid: (SEQ/64, 16 heads). Block: 128 thr = 4 waves.
// Wave owns 16 query rows; loops 32-key tiles to the diagonal. Online softmax
// via half-wave shuffles (C-layout rows live across 16 lanes). P re-layout
// C->A through a per-wave LDS tile ordered by s_wait_dscnt.
// Qh: [16][SEQ][256], Kh: [4][SEQ][256], Vt: [4][256][SEQ] (transposed),
// O: [NTOK][2048] token-major bf16.
// ---------------------------------------------------------------------------
__global__ __launch_bounds__(128)
void attn_kernel(const __bf16* __restrict__ Qh, const __bf16* __restrict__ Kh,
                 const __bf16* __restrict__ Vt, __bf16* __restrict__ O) {
    __shared__ __bf16 sP[4][16 * 32];
    int lane = threadIdx.x & 31;
    int wave = threadIdx.x >> 5;
    int hq = blockIdx.y;
    int qbase = blockIdx.x * 64 + wave * 16;
    int hk = (hq >> 3) * 2 + ((hq & 7) >> 2);

    const __bf16* Qb = Qh + (size_t)hq * SEQ * DIM;
    const __bf16* Kb = Kh + (size_t)hk * SEQ * DIM;
    const __bf16* Vb = Vt + (size_t)hk * DIM * SEQ;

    bf16x16 qf[8];
#pragma unroll
    for (int kk = 0; kk < 8; ++kk) qf[kk] = load_frag(Qb, qbase, DIM, kk * 32, lane);

    f32x8 zero = {};
    f32x8 o[16];
#pragma unroll
    for (int j = 0; j < 16; ++j) o[j] = zero;
    float mrow[8], lrow[8];
#pragma unroll
    for (int r = 0; r < 8; ++r) { mrow[r] = -3.0e38f; lrow[r] = 0.f; }

    int hf = lane >> 4, nn = lane & 15;
    int qlast = qbase + 15;
    const float scale = 0.0625f;   // 1/sqrt(256)

    for (int kb = 0; kb <= qlast; kb += 32) {
        if (kb + 32 <= qlast)                       // global_prefetch_b8
            __builtin_prefetch(Kb + (size_t)(kb + 32) * DIM, 0, 1);

        f32x8 s0 = zero, s1 = zero;                 // keys kb..+15, kb+16..+31
#pragma unroll
        for (int kk = 0; kk < 8; ++kk) {
            bf16x16 k0f = load_frag(Kb, kb,      DIM, kk * 32, lane);
            bf16x16 k1f = load_frag(Kb, kb + 16, DIM, kk * 32, lane);
            s0 = wmma_bf16(qf[kk], k0f, s0);
            s1 = wmma_bf16(qf[kk], k1f, s1);
        }
#pragma unroll
        for (int r = 0; r < 8; ++r) {
            int m = qbase + r + 8 * hf;
            float v0 = s0[r] * scale; if (kb + nn      > m) v0 = -3.0e38f;
            float v1 = s1[r] * scale; if (kb + 16 + nn > m) v1 = -3.0e38f;
            float tmax = fmaxf(v0, v1);
#pragma unroll
            for (int d = 1; d < 16; d <<= 1) tmax = fmaxf(tmax, __shfl_xor(tmax, d, 32));
            float mnew  = fmaxf(mrow[r], tmax);
            float alpha = __expf(mrow[r] - mnew);
            float p0 = __expf(v0 - mnew);
            float p1 = __expf(v1 - mnew);
            float lsum = p0 + p1;
#pragma unroll
            for (int d = 1; d < 16; d <<= 1) lsum += __shfl_xor(lsum, d, 32);
            lrow[r] = lrow[r] * alpha + lsum;
            mrow[r] = mnew;
#pragma unroll
            for (int j = 0; j < 16; ++j) o[j][r] *= alpha;
            int ml = r + 8 * hf;                    // local row 0..15
            sP[wave][ml * 32 + nn]      = (__bf16)p0;
            sP[wave][ml * 32 + 16 + nn] = (__bf16)p1;
        }
        asm volatile("s_wait_dscnt 0" ::: "memory"); // DS stores -> DS loads
        bf16x16 pf = load_frag(&sP[wave][0], 0, 32, 0, lane);
#pragma unroll
        for (int j = 0; j < 16; ++j) {
            bf16x16 vf = load_frag(Vb, j * 16, SEQ, kb, lane);
            o[j] = wmma_bf16(pf, vf, o[j]);
        }
    }

    int bb = hq >> 3, gu = hq & 7;
#pragma unroll
    for (int r = 0; r < 8; ++r) {
        float inv = 1.f / lrow[r];
        int srow = qbase + r + 8 * hf;
        size_t t = (size_t)srow * BSZ + bb;
#pragma unroll
        for (int j = 0; j < 16; ++j)
            O[t * 2048 + gu * DIM + j * 16 + nn] = (__bf16)(o[j][r] * inv);
    }
}

// ---------------------------------------------------------------------------
// Host orchestration
// ---------------------------------------------------------------------------
extern "C" void kernel_launch(void* const* d_in, const int* in_sizes, int n_in,
                              void* d_out, int out_size, void* d_ws, size_t ws_size,
                              hipStream_t stream) {
    const float* x   = (const float*)d_in[0];
    const float* g0  = (const float*)d_in[1];
    const float* b0  = (const float*)d_in[2];
    const float* g1  = (const float*)d_in[3];
    const float* b1  = (const float*)d_in[4];
    const float* gn  = (const float*)d_in[5];
    const float* bn  = (const float*)d_in[6];
    const float* wk  = (const float*)d_in[7];
    const float* bk  = (const float*)d_in[8];
    const float* wv  = (const float*)d_in[9];
    const float* bv  = (const float*)d_in[10];
    const float* wq  = (const float*)d_in[11];
    const float* bq  = (const float*)d_in[12];
    const float* wo  = (const float*)d_in[13];
    const float* bo  = (const float*)d_in[14];
    const float* w1  = (const float*)d_in[15];
    const float* bf1 = (const float*)d_in[16];
    const float* w2  = (const float*)d_in[17];
    const float* bf2 = (const float*)d_in[18];
    float* out = (float*)d_out;

    char* wsp = (char*)d_ws;
    auto carve = [&](size_t bytes) {
        char* p = wsp;
        wsp += (bytes + 255) & ~(size_t)255;
        return p;
    };
    __bf16* h_b  = (__bf16*)carve((size_t)NTOK * DIM * 2);
    __bf16* wk_b = (__bf16*)carve((size_t)512 * 256 * 2);
    __bf16* wv_b = (__bf16*)carve((size_t)512 * 256 * 2);
    __bf16* wq_b = (__bf16*)carve((size_t)2048 * 256 * 2);
    __bf16* wo_b = (__bf16*)carve((size_t)256 * 2048 * 2);
    __bf16* w1_b = (__bf16*)carve((size_t)512 * 256 * 2);
    __bf16* w2_b = (__bf16*)carve((size_t)256 * 512 * 2);
    __bf16* raw  = (__bf16*)carve((size_t)NTOK * 2048 * 2); // K/V/Q raw, reused
    __bf16* Kh   = (__bf16*)carve((size_t)4  * SEQ * DIM * 2);
    __bf16* Vt   = (__bf16*)carve((size_t)4  * DIM * SEQ * 2);
    __bf16* Qh   = (__bf16*)carve((size_t)16 * SEQ * DIM * 2);
    __bf16* Ob   = (__bf16*)carve((size_t)NTOK * 2048 * 2);
    float*  x2   = (float*)carve((size_t)NTOK * DIM * 4);
    __bf16* h2   = (__bf16*)carve((size_t)NTOK * DIM * 2);
    __bf16* F1   = (__bf16*)carve((size_t)NTOK * 512 * 2);

    auto cvt = [&](const float* src, __bf16* dst, int n) {
        cvt_bf16<<<(n + 255) / 256, 256, 0, stream>>>(src, dst, n);
    };
    cvt(wk, wk_b, 512 * 256);
    cvt(wv, wv_b, 512 * 256);
    cvt(wq, wq_b, 2048 * 256);
    cvt(wo, wo_b, 256 * 2048);
    cvt(w1, w1_b, 512 * 256);
    cvt(w2, w2_b, 256 * 512);

    // h = LN(x; g0,b0)
    ln_rows<<<NTOK, 256, 0, stream>>>(x, h_b, g0, b0);

    auto gemm = [&](const __bf16* A, int lda, const __bf16* W, int ldw,
                    const float* bias, const float* resid, float* Cf, __bf16* Cb,
                    int M, int N, int K, int relu) {
        dim3 grid(N / 64, M / 128);
        gemm_wmma<<<grid, 256, 0, stream>>>(A, lda, W, ldw, bias, resid, Cf, Cb,
                                            M, N, K, relu);
    };

    // K projection + per-head LN -> Kh[4][SEQ][256]
    gemm(h_b, 256, wk_b, 256, bk, nullptr, nullptr, raw, NTOK, 512, 256, 0);
    heads_postln<<<NTOK * GK, 256, 0, stream>>>(raw, Kh, gn, bn, GK, 1, 0);
    // V projection (no LN) -> transposed Vt[4][256][SEQ]
    gemm(h_b, 256, wv_b, 256, bv, nullptr, nullptr, raw, NTOK, 512, 256, 0);
    heads_postln<<<NTOK * GK, 256, 0, stream>>>(raw, Vt, gn, bn, GK, 0, 1);
    // Q projection + per-head LN -> Qh[16][SEQ][256]
    gemm(h_b, 256, wq_b, 256, bq, nullptr, nullptr, raw, NTOK, 2048, 256, 0);
    heads_postln<<<NTOK * NQH, 256, 0, stream>>>(raw, Qh, gn, bn, NQH, 1, 0);

    // causal flash attention -> Ob[NTOK][2048]
    attn_kernel<<<dim3(SEQ / 64, 16), 128, 0, stream>>>(Qh, Kh, Vt, Ob);

    // x2 = x + Ob @ wo^T + bo
    gemm(Ob, 2048, wo_b, 2048, bo, x, x2, nullptr, NTOK, 256, 2048, 0);
    // h2 = LN(x2; g1,b1)
    ln_rows<<<NTOK, 256, 0, stream>>>(x2, h2, g1, b1);
    // F1 = relu(h2 @ w1^T + bf1)
    gemm(h2, 256, w1_b, 256, bf1, nullptr, nullptr, F1, NTOK, 512, 256, 1);
    // out = x2 + F1 @ w2^T + bf2
    gemm(F1, 512, w2_b, 512, bf2, x2, out, nullptr, NTOK, 256, 512, 0);

    (void)in_sizes; (void)n_in; (void)out_size; (void)ws_size;
}
